// SlowMultiHeadAttention_27633819582880
// MI455X (gfx1250) — compile-verified, matrix operand order checked by reading, not run on hardware
//
#include <hip/hip_runtime.h>
#include <hip/hip_bf16.h>
#include <math.h>

// Problem constants (from reference)
#define BB   4
#define SS   2048
#define HH   16
#define DD   64
#define OUTD 1024
#define HD   (HH * DD)   // 1024

typedef __attribute__((ext_vector_type(16))) __bf16 v16bf;
typedef __attribute__((ext_vector_type(8)))  __bf16 v8bf;
typedef __attribute__((ext_vector_type(8)))  float  v8f;

static __device__ __forceinline__ v8f wmma_bf16(v16bf a, v16bf b, v8f c) {
  // D = A(16x32 bf16) x B(32x16 bf16) + C(16x16 f32)
  return __builtin_amdgcn_wmma_f32_16x16x32_bf16(
      /*neg_a=*/false, a, /*neg_b=*/false, b,
      /*c_mod=*/(short)0, c, /*reuse_a=*/false, /*reuse_b=*/false);
}

// Cross-lane xor-exchange via DS_SWIZZLE_B32 (group-of-32 mode):
// dst lane i gets src lane (i ^ M); masks <16 stay within each 16-lane half.
template <int M>
static __device__ __forceinline__ float xor_swz(float v) {
  return __int_as_float(
      __builtin_amdgcn_ds_swizzle(__float_as_int(v), (M << 10) | 0x1f));
}

// ---- Fragment loaders ------------------------------------------------------
// 16-bit A-matrix 16x32 layout (ISA 7.12.2):
//   lane<16 : row = lane,    elems 0..7 = K 0..7,  elems 8..15 = K 16..23
//   lane>=16: row = lane-16, elems 0..7 = K 8..15, elems 8..15 = K 24..31
// B mirrors A with M<->N; a column-of-B contiguous in K loads identically.

static __device__ __forceinline__ v16bf load_frag_bf16(const __bf16* p, int ld) {
  const int lane = threadIdx.x & 31;
  const int idx  = lane & 15;
  const int half = lane >> 4;
  const __bf16* r = p + (size_t)idx * ld + half * 8;
  v8bf lo = *reinterpret_cast<const v8bf*>(r);        // K base+0..7
  v8bf hi = *reinterpret_cast<const v8bf*>(r + 16);   // K base+16..23
  return __builtin_shufflevector(lo, hi, 0, 1, 2, 3, 4, 5, 6, 7,
                                 8, 9, 10, 11, 12, 13, 14, 15);
}

static __device__ __forceinline__ v16bf load_frag_f32(const float* p, int ld) {
  const int lane = threadIdx.x & 31;
  const int idx  = lane & 15;
  const int half = lane >> 4;
  const float* r = p + (size_t)idx * ld + half * 8;
  v16bf out;
#pragma unroll
  for (int i = 0; i < 8; ++i) {
    out[i]     = (__bf16)r[i];
    out[i + 8] = (__bf16)r[16 + i];
  }
  return out;
}

// Online-softmax max/exp step for one 16x32 score tile in C-layout.
// Computes new running max m, rescale factor alpha, and probabilities p0/p1.
// (Row sums are computed later with a WMMA against a ones-matrix.)
static __device__ __forceinline__ void softmax_max_step(
    const v8f& s0, const v8f& s1, float m[8],
    float p0[8], float p1[8], float alpha[8], float cexp) {
#pragma unroll
  for (int r = 0; r < 8; ++r) {
    float mx = fmaxf(s0[r], s1[r]);
    mx = fmaxf(mx, xor_swz<1>(mx));
    mx = fmaxf(mx, xor_swz<2>(mx));
    mx = fmaxf(mx, xor_swz<4>(mx));
    mx = fmaxf(mx, xor_swz<8>(mx));
    const float mn = fmaxf(m[r], mx);
    alpha[r] = __builtin_exp2f((m[r] - mn) * cexp);
    p0[r]    = __builtin_exp2f((s0[r] - mn) * cexp);
    p1[r]    = __builtin_exp2f((s1[r] - mn) * cexp);
    m[r] = mn;
  }
}

// ---- Kernel 0: fp32 -> bf16 convert (for Wo) -------------------------------
__global__ void cvt_bf16_kernel(const float* __restrict__ in,
                                __bf16* __restrict__ out, int n) {
  int i = blockIdx.x * blockDim.x + threadIdx.x;
  if (i < n) out[i] = (__bf16)in[i];
}

// ---- Kernel 1: per-head projection  out[b,h,s,e] = sum_d X[b,s,d]*W[h,e,d]
// VT=false -> row-major [B,H,S,D];  VT=true -> transposed [B,H,D,S] (for V).
template <bool VT>
__global__ __launch_bounds__(32)
void proj_kernel(const float* __restrict__ X, const float* __restrict__ W,
                 __bf16* __restrict__ Out) {
  const int st   = blockIdx.x;           // S/16 row tiles
  const int h    = blockIdx.y;
  const int b    = blockIdx.z;
  const int lane = threadIdx.x & 31;
  const int idx  = lane & 15;
  const int half = lane >> 4;
  const int s0   = st * 16;

  v8f acc[4] = {};
#pragma unroll
  for (int k0 = 0; k0 < DD; k0 += 32) {
    v16bf a = load_frag_f32(X + ((size_t)b * SS + s0) * DD + k0, DD);
#pragma unroll
    for (int t = 0; t < 4; ++t) {
      v16bf bf = load_frag_f32(W + ((size_t)h * DD + t * 16) * DD + k0, DD);
      acc[t] = wmma_bf16(a, bf, acc[t]);
    }
  }
  if (!VT) {
    __bf16* o = Out + (((size_t)b * HH + h) * SS + s0) * DD;
#pragma unroll
    for (int t = 0; t < 4; ++t)
#pragma unroll
      for (int r = 0; r < 8; ++r)
        o[(size_t)(r + 8 * half) * DD + t * 16 + idx] = (__bf16)acc[t][r];
  } else {
    __bf16* o = Out + ((size_t)b * HH + h) * (size_t)DD * SS;
#pragma unroll
    for (int t = 0; t < 4; ++t)
#pragma unroll
      for (int r = 0; r < 8; ++r)
        o[(size_t)(t * 16 + idx) * SS + s0 + r + 8 * half] = (__bf16)acc[t][r];
  }
}

// ---- Kernel 2: flash attention (32 query rows per wave) --------------------
__global__ __launch_bounds__(32)
void attn_kernel(const __bf16* __restrict__ Q, const __bf16* __restrict__ K,
                 const __bf16* __restrict__ Vt, __bf16* __restrict__ O) {
  __shared__ __align__(16) __bf16 Kb[32 * DD];   // 4 KB K-tile (async staged)
  __shared__ __align__(16) __bf16 Pb[32 * 32];   // 2 KB P staging (C -> A layout)

  const int qt   = blockIdx.x;                   // S/32 tiles
  const int h    = blockIdx.y;
  const int b    = blockIdx.z;
  const int lane = threadIdx.x & 31;
  const int idx  = lane & 15;
  const int half = lane >> 4;
  const size_t bh = ((size_t)b * HH + h) * SS;

  const __bf16* qp = Q + (bh + qt * 32) * DD;
  const v16bf qA0 = load_frag_bf16(qp, DD);                 // rows 0-15, K 0..31
  const v16bf qA1 = load_frag_bf16(qp + 32, DD);            // rows 0-15, K 32..63
  const v16bf qB0 = load_frag_bf16(qp + 16 * DD, DD);       // rows 16-31
  const v16bf qB1 = load_frag_bf16(qp + 16 * DD + 32, DD);

  v16bf onesf;
#pragma unroll
  for (int i = 0; i < 16; ++i) onesf[i] = (__bf16)1.0f;

  float mA[8], lA[8], mB[8], lB[8];
  v8f oA[4] = {}, oB[4] = {};
#pragma unroll
  for (int r = 0; r < 8; ++r) {
    mA[r] = mB[r] = -1e30f;
    lA[r] = lB[r] = 0.f;
  }
  const float cexp = 0.125f * 1.44269504f;   // (1/sqrt(D)) * log2(e)

  // LDS byte address of K staging buffer (flat-aperture low 32 bits == LDS offset)
  const unsigned kb_lane = (unsigned)(size_t)&Kb[0] + (unsigned)(lane * 16);
  const __bf16* kl = &Kb[0];
  const __bf16* vbase = Vt + ((size_t)b * HH + h) * (size_t)DD * SS;

  for (int kt = 0; kt < SS; kt += 32) {
    // ---- async stage 32x64 bf16 K tile into LDS (8 x 512B, all 32 lanes) ----
    const unsigned long long gk =
        (unsigned long long)(size_t)(K + (bh + kt) * DD) + (unsigned)(lane * 16);
#pragma unroll
    for (int i = 0; i < 8; ++i)
      asm volatile("global_load_async_to_lds_b128 %0, %1, off"
                   :: "v"(kb_lane + i * 512), "v"(gk + i * 512) : "memory");
    __builtin_prefetch(K + (bh + kt + 32) * DD, 0, 0);   // next tile -> L2
    asm volatile("s_wait_asynccnt 0" ::: "memory");

    // ---- scores: two 16-row q tiles share every K fragment ----
    v8f sA0 = {}, sA1 = {}, sB0 = {}, sB1 = {};
    v16bf kf;
    kf = load_frag_bf16(kl, DD);
    sA0 = wmma_bf16(qA0, kf, sA0);  sB0 = wmma_bf16(qB0, kf, sB0);
    kf = load_frag_bf16(kl + 32, DD);
    sA0 = wmma_bf16(qA1, kf, sA0);  sB0 = wmma_bf16(qB1, kf, sB0);
    kf = load_frag_bf16(kl + 16 * DD, DD);
    sA1 = wmma_bf16(qA0, kf, sA1);  sB1 = wmma_bf16(qB0, kf, sB1);
    kf = load_frag_bf16(kl + 16 * DD + 32, DD);
    sA1 = wmma_bf16(qA1, kf, sA1);  sB1 = wmma_bf16(qB1, kf, sB1);

    // ---- online softmax: running max + probabilities ----
    float pA0[8], pA1[8], aAl[8], pB0[8], pB1[8], aBl[8];
    softmax_max_step(sA0, sA1, mA, pA0, pA1, aAl, cexp);
    softmax_max_step(sB0, sB1, mB, pB0, pB1, aBl, cexp);
#pragma unroll
    for (int t = 0; t < 4; ++t)
#pragma unroll
      for (int r = 0; r < 8; ++r) {
        oA[t][r] *= aAl[r];
        oB[t][r] *= aBl[r];
      }

    // ---- stage P (32x32) through LDS: C-layout -> A-fragment layout ----
#pragma unroll
    for (int r = 0; r < 8; ++r) {
      Pb[(r + 8 * half) * 32 + idx]             = (__bf16)pA0[r];
      Pb[(r + 8 * half) * 32 + 16 + idx]        = (__bf16)pA1[r];
      Pb[(16 + r + 8 * half) * 32 + idx]        = (__bf16)pB0[r];
      Pb[(16 + r + 8 * half) * 32 + 16 + idx]   = (__bf16)pB1[r];
    }
    asm volatile("s_wait_dscnt 0" ::: "memory");
    const v16bf pfA = load_frag_bf16(&Pb[0], 32);
    const v16bf pfB = load_frag_bf16(&Pb[16 * 32], 32);

    // ---- row sums of P via WMMA against ones (every lane gets its row sum,
    //      and sums exactly the bf16-quantized P used in the P*V numerator) ----
    v8f z = {};
    const v8f rsA = wmma_bf16(pfA, onesf, z);
    const v8f rsB = wmma_bf16(pfB, onesf, z);
#pragma unroll
    for (int r = 0; r < 8; ++r) {
      lA[r] = lA[r] * aAl[r] + rsA[r];
      lB[r] = lB[r] * aBl[r] + rsB[r];
    }

    // ---- P*V: V^T columns contiguous in key index, shared by both tiles ----
#pragma unroll
    for (int t = 0; t < 4; ++t) {
      const v16bf vf = load_frag_bf16(vbase + (size_t)(t * 16) * SS + kt, SS);
      oA[t] = wmma_bf16(pfA, vf, oA[t]);
      oB[t] = wmma_bf16(pfB, vf, oB[t]);
    }
  }

  // ---- normalize + write concat-head layout [B, S, H*D] ----
  __bf16* op = O + ((size_t)b * SS + qt * 32) * HD + h * DD;
#pragma unroll
  for (int r = 0; r < 8; ++r) {
    const float rlA = 1.f / lA[r];
    const float rlB = 1.f / lB[r];
#pragma unroll
    for (int t = 0; t < 4; ++t) {
      op[(size_t)(r + 8 * half) * HD + t * 16 + idx]      = (__bf16)(oA[t][r] * rlA);
      op[(size_t)(16 + r + 8 * half) * HD + t * 16 + idx] = (__bf16)(oB[t][r] * rlB);
    }
  }
}

// ---- Kernel 3: output projection  Y = X[8192,1024] * Wo^T ------------------
__global__ __launch_bounds__(32)
void outproj_kernel(const __bf16* __restrict__ X, const __bf16* __restrict__ Wo,
                    float* __restrict__ Y) {
  const int it   = blockIdx.x;       // 256 row tiles of 32
  const int ot   = blockIdx.y;       // 16 col groups of 64
  const int lane = threadIdx.x & 31;
  const int idx  = lane & 15;
  const int half = lane >> 4;
  const int i0   = it * 32;
  const int o0   = ot * 64;

  v8f accA[4] = {}, accB[4] = {};
  for (int c0 = 0; c0 < HD; c0 += 32) {
    __builtin_prefetch(X + (size_t)i0 * HD + c0 + 64, 0, 0);
    v16bf aA = load_frag_bf16(X + (size_t)i0 * HD + c0, HD);
    v16bf aB = load_frag_bf16(X + (size_t)(i0 + 16) * HD + c0, HD);
#pragma unroll
    for (int t = 0; t < 4; ++t) {
      v16bf bf = load_frag_bf16(Wo + (size_t)(o0 + t * 16) * HD + c0, HD);
      accA[t] = wmma_bf16(aA, bf, accA[t]);
      accB[t] = wmma_bf16(aB, bf, accB[t]);
    }
  }
#pragma unroll
  for (int t = 0; t < 4; ++t)
#pragma unroll
    for (int r = 0; r < 8; ++r) {
      Y[(size_t)(i0 + r + 8 * half) * OUTD + o0 + t * 16 + idx]      = accA[t][r];
      Y[(size_t)(i0 + 16 + r + 8 * half) * OUTD + o0 + t * 16 + idx] = accB[t][r];
    }
}

// ---- Launch ----------------------------------------------------------------
extern "C" void kernel_launch(void* const* d_in, const int* in_sizes, int n_in,
                              void* d_out, int out_size, void* d_ws, size_t ws_size,
                              hipStream_t stream) {
  const float* q  = (const float*)d_in[0];
  const float* k  = (const float*)d_in[1];
  const float* v  = (const float*)d_in[2];
  const float* Wq = (const float*)d_in[3];
  const float* Wk = (const float*)d_in[4];
  const float* Wv = (const float*)d_in[5];
  const float* Wo = (const float*)d_in[6];
  float* out = (float*)d_out;

  // Workspace (bf16): q,k [B,H,S,D]; v^T [B,H,D,S]; attn-out [B,S,H*D]; Wo
  __bf16* ws = (__bf16*)d_ws;
  const size_t NE = (size_t)BB * HH * SS * DD;   // 8M elems per tensor
  __bf16* qb  = ws;
  __bf16* kb  = ws + NE;
  __bf16* vtb = ws + 2 * NE;
  __bf16* ob  = ws + 3 * NE;
  __bf16* wob = ws + 4 * NE;

  dim3 blk(32);
  dim3 gproj(SS / 16, HH, BB);                   // 128 x 16 x 4

  proj_kernel<false><<<gproj, blk, 0, stream>>>(q, Wq, qb);
  proj_kernel<false><<<gproj, blk, 0, stream>>>(k, Wk, kb);
  proj_kernel<true ><<<gproj, blk, 0, stream>>>(v, Wv, vtb);
  cvt_bf16_kernel<<<(OUTD * HD + 255) / 256, 256, 0, stream>>>(Wo, wob, OUTD * HD);

  dim3 gattn(SS / 32, HH, BB);                   // 64 x 16 x 4
  attn_kernel<<<gattn, blk, 0, stream>>>(qb, kb, vtb, ob);

  dim3 gout(BB * SS / 32, OUTD / 64);            // 256 x 16
  outproj_kernel<<<gout, blk, 0, stream>>>(ob, wob, out);
}
